// rnn_no_cleanup_46514495816224
// MI455X (gfx1250) — matrix operand order repaired and checked by reading, not compiled, not run on hardware
//
#include <hip/hip_runtime.h>
#include <hip/hip_bf16.h>
#include <stdint.h>

typedef __attribute__((ext_vector_type(16))) __bf16 v16bf;
typedef __attribute__((ext_vector_type(8)))  float  v8f;
typedef __attribute__((ext_vector_type(4)))  float  v4f;

#define BATCH  8192
#define TSTEPS 12
#define IN_DIM 364
#define IN_PAD 384
#define HID    512
#define OUTD   250
#define OUTP   256
#define TAU    0.2f

#define KT_OH (IN_PAD / 32)  // 12 K-tiles for X @ w_oh
#define NT_OH (HID / 16)     // 32 N-tiles
#define KT_HP (HID / 32)     // 16 K-tiles for sig(h) @ w_hp
#define NT_HP (OUTP / 16)    // 16 N-tiles
#define KT_PP (OUTP / 32)    // 8  K-tiles for sig(p) @ w_pp
#define NK2   (KT_HP + KT_PP)// 24 fused K-tiles for GEMM2 (contiguous LDS strip)

#define BLOCK_THREADS 128    // 4 waves; each wave owns 16 batch rows
#define FRAG_BYTES 1024      // one 32x16 bf16 B-fragment: 32 lanes x 32B

__device__ __forceinline__ float sigm(float x) {
  return __builtin_amdgcn_rcpf(1.0f + __expf(-x));
}

// Async global->LDS copy of 16B per lane (ASYNCcnt-tracked).
// Probe-derived signature: (AS1 int4* src, AS3 int4* dst, imm offset, imm cpol).
typedef int b128_t __attribute__((vector_size(16)));
typedef __attribute__((address_space(1))) b128_t* gas_b128_p;
typedef __attribute__((address_space(3))) b128_t* las_b128_p;

__device__ __forceinline__ void async_b128(const void* g, void* l) {
#if __has_builtin(__builtin_amdgcn_global_load_async_to_lds_b128)
  __builtin_amdgcn_global_load_async_to_lds_b128(
      (gas_b128_p)(uintptr_t)g, (las_b128_p)(uintptr_t)l, 0, 0);
#else
  unsigned lds = (unsigned)(uintptr_t)l;  // generic addr low 32 bits = LDS byte addr
  unsigned long long ga = (unsigned long long)(uintptr_t)g;
  asm volatile("global_load_async_to_lds_b128 %0, %1, off"
               :: "v"(lds), "v"(ga) : "memory");
#endif
}

__device__ __forceinline__ void wait_async0() {
#if __has_builtin(__builtin_amdgcn_s_wait_asynccnt)
  __builtin_amdgcn_s_wait_asynccnt(0);
#else
  asm volatile("s_wait_asynccnt 0" ::: "memory");
#endif
}

// ---------------------------------------------------------------------------
// Weight pre-pack: f32 row-major [K x N] -> bf16 fragment-linear layout,
// *nt-major* (frag = nt*KT + kt) so a per-nt strip is one contiguous block
// for the async LDS staging. Lane l of the consuming wave needs column
// n = nt*16 + (l&15), K-locals (l>>4)*16 + e, stored contiguously (32B/lane).
// ---------------------------------------------------------------------------
__global__ void pack_oh_kernel(const float* __restrict__ w, __bf16* __restrict__ dst) {
  int idx = blockIdx.x * blockDim.x + threadIdx.x;
  if (idx >= KT_OH * NT_OH * 32) return;
  int lane = idx & 31, frag = idx >> 5;
  int nt = frag / KT_OH, kt = frag % KT_OH;
  int n  = nt * 16 + (lane & 15);
  int kb = kt * 32 + (lane >> 4) * 16;
#pragma unroll
  for (int e = 0; e < 16; ++e) {
    int k = kb + e;
    float v = (k < IN_DIM) ? w[(size_t)k * HID + n] : 0.0f;  // zero-pad K 364->384
    dst[(size_t)idx * 16 + e] = (__bf16)v;
  }
}

__global__ void pack_hp_kernel(const float* __restrict__ w, __bf16* __restrict__ dst) {
  int idx = blockIdx.x * blockDim.x + threadIdx.x;
  if (idx >= KT_HP * NT_HP * 32) return;
  int lane = idx & 31, frag = idx >> 5;
  int nt = frag / KT_HP, kt = frag % KT_HP;
  int n  = nt * 16 + (lane & 15);
  int kb = kt * 32 + (lane >> 4) * 16;
#pragma unroll
  for (int e = 0; e < 16; ++e) {
    int k = kb + e;
    float v = (n < OUTD) ? w[(size_t)k * OUTD + n] : 0.0f;   // zero-pad N 250->256
    dst[(size_t)idx * 16 + e] = (__bf16)v;
  }
}

__global__ void pack_pp_kernel(const float* __restrict__ w, __bf16* __restrict__ dst) {
  int idx = blockIdx.x * blockDim.x + threadIdx.x;
  if (idx >= KT_PP * NT_HP * 32) return;
  int lane = idx & 31, frag = idx >> 5;
  int nt = frag / KT_PP, kt = frag % KT_PP;
  int n  = nt * 16 + (lane & 15);
  int kb = kt * 32 + (lane >> 4) * 16;
#pragma unroll
  for (int e = 0; e < 16; ++e) {
    int k = kb + e;
    float v = (k < OUTD && n < OUTD) ? w[(size_t)k * OUTD + n] : 0.0f;  // pad K & N
    dst[(size_t)idx * 16 + e] = (__bf16)v;
  }
}

__global__ void zero_f32_kernel(float* __restrict__ p, int n) {
  for (int i = blockIdx.x * blockDim.x + threadIdx.x; i < n; i += gridDim.x * blockDim.x)
    p[i] = 0.0f;
}

// ---------------------------------------------------------------------------
// Fused RNN. Each wave32 owns 16 batch rows for all 12 timesteps (rows are
// independent -> no cross-wave state sharing). All 4 waves consume identical
// weight fragments, so the block double-buffers per-nt B strips in LDS via
// async global->LDS DMA, overlapped with WMMA compute on the previous strip.
// Inside a strip, B fragments are consumed in double-buffered chunks of 4
// with two independent accumulator chains, so ds_load latency overlaps WMMA
// issue and the WMMA->WMMA acc RAW chain is broken.
// ---------------------------------------------------------------------------
__global__ __launch_bounds__(BLOCK_THREADS) void rnn_fused_kernel(
    const float* __restrict__ x,        // [BATCH, TSTEPS, IN_DIM]
    const float* __restrict__ bias_h,   // [HID]
    const float* __restrict__ bias_p,   // [OUTD]
    const __bf16* __restrict__ pOH, const __bf16* __restrict__ pHP,
    const __bf16* __restrict__ pPP,
    float* __restrict__ h0, float* __restrict__ h1,   // [BATCH, HID]
    float* __restrict__ p0, float* __restrict__ p1,   // [BATCH, OUTP]
    float* __restrict__ out)                          // [TSTEPS, BATCH, OUTD]
{
  __shared__ __align__(32) char ldsB[2][NK2 * FRAG_BYTES];  // 2 x 24KB

  const int tid   = threadIdx.x;
  const int lane  = tid & 31;
  const int wave  = tid >> 5;
  const int rbase = (blockIdx.x * 4 + wave) * 16;
  const int nlow  = lane & 15;
  const int hi    = lane >> 4;
  const int arow  = rbase + nlow;  // this lane's A-fragment row

  // Stage the B strip for output tile `nt` of GEMM2 (w_hp then w_pp) into buf b.
  auto stage2 = [&](int nt, int b) {
    const char* gh = (const char*)pHP + (size_t)nt * (KT_HP * FRAG_BYTES);
    const char* gp = (const char*)pPP + (size_t)nt * (KT_PP * FRAG_BYTES);
    char* L = ldsB[b];
#pragma unroll
    for (int i = tid; i < KT_HP * 64; i += BLOCK_THREADS)          // 16KB
      async_b128(gh + (size_t)i * 16, L + (size_t)i * 16);
#pragma unroll
    for (int i = tid; i < KT_PP * 64; i += BLOCK_THREADS)          // 8KB
      async_b128(gp + (size_t)i * 16, L + KT_HP * FRAG_BYTES + (size_t)i * 16);
  };
  // Stage the B strip for output tile `nt` of GEMM1 (w_oh) into buf b.
  auto stage1 = [&](int nt, int b) {
    const char* g = (const char*)pOH + (size_t)nt * (KT_OH * FRAG_BYTES);
    char* L = ldsB[b];
#pragma unroll
    for (int i = tid; i < KT_OH * 64; i += BLOCK_THREADS)          // 12KB
      async_b128(g + (size_t)i * 16, L + (size_t)i * 16);
  };

  for (int t = 0; t < TSTEPS; ++t) {
    const float* hPrev = (t & 1) ? h1 : h0;
    float*       hNext = (t & 1) ? h0 : h1;
    const float* pPrev = (t & 1) ? p1 : p0;
    float*       pNext = (t & 1) ? p0 : p1;

    if (t + 1 < TSTEPS)
      __builtin_prefetch(x + ((size_t)arow * TSTEPS + t + 1) * IN_DIM, 0, 3);

    // ---- hoisted A fragments: sigmoid of previous states (computed once) ----
    // aF[0..15]  = sig(h_prev) K-tiles, aF[16..23] = sig(p_prev) K-tiles.
    // k order used here (contiguous): lanes 0-15 hold K {0..15}, lanes 16-31 {16..31}
    v16bf aF[NK2];
    {
      const float* hrow = hPrev + (size_t)arow * HID;
#pragma unroll
      for (int kt = 0; kt < KT_HP; ++kt) {
        int b1 = kt * 32 + hi * 16, b2 = b1 + 8;
        v4f a = *(const v4f*)(hrow + b1);
        v4f b = *(const v4f*)(hrow + b1 + 4);
        v4f c = *(const v4f*)(hrow + b2);
        v4f d = *(const v4f*)(hrow + b2 + 4);
        v16bf r;
#pragma unroll
        for (int j = 0; j < 4; ++j) {
          r[j]      = (__bf16)sigm(a[j]);
          r[4 + j]  = (__bf16)sigm(b[j]);
          r[8 + j]  = (__bf16)sigm(c[j]);
          r[12 + j] = (__bf16)sigm(d[j]);
        }
        aF[kt] = r;
      }
      const float* prow = pPrev + (size_t)arow * OUTP;
#pragma unroll
      for (int kt = 0; kt < KT_PP; ++kt) {
        int b1 = kt * 32 + hi * 16, b2 = b1 + 8;
        v4f a = *(const v4f*)(prow + b1);
        v4f b = *(const v4f*)(prow + b1 + 4);
        v4f c = *(const v4f*)(prow + b2);
        v4f d = *(const v4f*)(prow + b2 + 4);
        v16bf r;
#pragma unroll
        for (int j = 0; j < 4; ++j) {
          r[j]      = (__bf16)sigm(a[j]);
          r[4 + j]  = (__bf16)sigm(b[j]);
          r[8 + j]  = (__bf16)sigm(c[j]);
          r[12 + j] = (__bf16)sigm(d[j]);
        }
        aF[KT_HP + kt] = r;
      }
    }

    // ---- GEMM2: p_new = tau*(sig(h)@Whp + sig(p)@Wpp + bp) + 0.8*p_prev ----
    stage2(0, 0);
    for (int nt = 0; nt < NT_HP; ++nt) {
      const int cur = nt & 1;
      wait_async0();          // my share of strip `nt` landed in LDS
      __syncthreads();        // everyone's share landed; prior readers of buf cur^1 done
      if (nt + 1 < NT_HP) stage2(nt + 1, cur ^ 1);   // DMA next strip during compute

      const char* L = ldsB[cur];
      auto frag = [&](int kt) -> v16bf {
        return *(const v16bf*)(L + (size_t)(kt * 32 + lane) * 32);
      };
      // Double-buffered chunks of 4 B-fragments + dual acc chains.
      v8f acc0 = {}, acc1 = {};
      v16bf buf0[4], buf1[4];
#pragma unroll
      for (int i = 0; i < 4; ++i) buf0[i] = frag(i);
#pragma unroll
      for (int c = 0; c < NK2 / 4; ++c) {            // 6 chunks
        v16bf* curb = (c & 1) ? buf1 : buf0;
        v16bf* nxtb = (c & 1) ? buf0 : buf1;
        if (c + 1 < NK2 / 4) {
#pragma unroll
          for (int i = 0; i < 4; ++i) nxtb[i] = frag(4 * (c + 1) + i);
        }
#pragma unroll
        for (int i = 0; i < 4; ++i) {
          if (i & 1)
            acc1 = __builtin_amdgcn_wmma_f32_16x16x32_bf16(
                false, aF[4 * c + i], false, curb[i], (short)0, acc1, false, false);
          else
            acc0 = __builtin_amdgcn_wmma_f32_16x16x32_bf16(
                false, aF[4 * c + i], false, curb[i], (short)0, acc0, false, false);
        }
      }
      v8f acc = acc0 + acc1;

      int col = nt * 16 + nlow;
      float bp = (col < OUTD) ? bias_p[col] : 0.0f;
#pragma unroll
      for (int g = 0; g < 8; ++g) {  // C/D layout: M = g + 8*hi, N = lane&15
        int row = rbase + g + 8 * hi;
        float pv = TAU * (acc[g] + bp) + (1.0f - TAU) * pPrev[(size_t)row * OUTP + col];
        pNext[(size_t)row * OUTP + col] = pv;
        if (col < OUTD)
          out[((size_t)t * BATCH + row) * OUTD + col] = sigm(pv);
      }
    }

    // ---- X_t fragments (after GEMM2 to keep peak VGPR liveness low) ----
    v16bf xF[KT_OH];
    {
      const float* xrow = x + ((size_t)arow * TSTEPS + t) * IN_DIM;
#pragma unroll
      for (int kt = 0; kt < KT_OH; ++kt) {
        int b1 = kt * 32 + hi * 16, b2 = b1 + 8;
        v16bf r;
        if (kt < KT_OH - 1) {  // fully in range: vector loads (rows 16B aligned)
          v4f a = *(const v4f*)(xrow + b1);
          v4f b = *(const v4f*)(xrow + b1 + 4);
          v4f c = *(const v4f*)(xrow + b2);
          v4f d = *(const v4f*)(xrow + b2 + 4);
#pragma unroll
          for (int j = 0; j < 4; ++j) {
            r[j] = (__bf16)a[j]; r[4 + j] = (__bf16)b[j];
            r[8 + j] = (__bf16)c[j]; r[12 + j] = (__bf16)d[j];
          }
        } else {               // tail K-tile: guard k < 364, zero-pad the rest
#pragma unroll
          for (int j = 0; j < 8; ++j) {
            int k1 = b1 + j, k2 = b2 + j;
            r[j]     = (__bf16)((k1 < IN_DIM) ? xrow[k1] : 0.0f);
            r[8 + j] = (__bf16)((k2 < IN_DIM) ? xrow[k2] : 0.0f);
          }
        }
        xF[kt] = r;
      }
    }

    // ---- GEMM1: h_new = tau*(X_t@Woh + bh) + 0.8*h_prev ----
    stage1(0, 0);
    for (int nt = 0; nt < NT_OH; ++nt) {
      const int cur = nt & 1;
      wait_async0();
      __syncthreads();
      if (nt + 1 < NT_OH) stage1(nt + 1, cur ^ 1);

      const char* L = ldsB[cur];
      auto frag = [&](int kt) -> v16bf {
        return *(const v16bf*)(L + (size_t)(kt * 32 + lane) * 32);
      };
      v8f acc0 = {}, acc1 = {};
      v16bf buf0[4], buf1[4];
#pragma unroll
      for (int i = 0; i < 4; ++i) buf0[i] = frag(i);
#pragma unroll
      for (int c = 0; c < KT_OH / 4; ++c) {          // 3 chunks
        v16bf* curb = (c & 1) ? buf1 : buf0;
        v16bf* nxtb = (c & 1) ? buf0 : buf1;
        if (c + 1 < KT_OH / 4) {
#pragma unroll
          for (int i = 0; i < 4; ++i) nxtb[i] = frag(4 * (c + 1) + i);
        }
#pragma unroll
        for (int i = 0; i < 4; ++i) {
          if (i & 1)
            acc1 = __builtin_amdgcn_wmma_f32_16x16x32_bf16(
                false, xF[4 * c + i], false, curb[i], (short)0, acc1, false, false);
          else
            acc0 = __builtin_amdgcn_wmma_f32_16x16x32_bf16(
                false, xF[4 * c + i], false, curb[i], (short)0, acc0, false, false);
        }
      }
      v8f acc = acc0 + acc1;

      int col = nt * 16 + nlow;
      float bh = bias_h[col];
#pragma unroll
      for (int g = 0; g < 8; ++g) {
        int row = rbase + g + 8 * hi;
        hNext[(size_t)row * HID + col] =
            TAU * (acc[g] + bh) + (1.0f - TAU) * hPrev[(size_t)row * HID + col];
      }
    }
  }
}

// ---------------------------------------------------------------------------
extern "C" void kernel_launch(void* const* d_in, const int* in_sizes, int n_in,
                              void* d_out, int out_size, void* d_ws, size_t ws_size,
                              hipStream_t stream) {
  const float* x      = (const float*)d_in[0];  // [8192,12,364]
  const float* w_oh   = (const float*)d_in[1];  // [364,512]
  const float* w_hp   = (const float*)d_in[2];  // [512,250]
  const float* w_pp   = (const float*)d_in[3];  // [250,250]
  const float* bias_h = (const float*)d_in[4];  // [512]
  const float* bias_p = (const float*)d_in[5];  // [250]
  float* out = (float*)d_out;                   // [12,8192,250]

  // Workspace layout: packed bf16 weights (nt-major), then ping-pong f32 states.
  char* ws = (char*)d_ws;
  const size_t szOH = (size_t)KT_OH * NT_OH * 32 * 16 * sizeof(__bf16);  // 384 KB
  const size_t szHP = (size_t)KT_HP * NT_HP * 32 * 16 * sizeof(__bf16);  // 256 KB
  const size_t szPP = (size_t)KT_PP * NT_HP * 32 * 16 * sizeof(__bf16);  // 128 KB
  const size_t szH  = (size_t)BATCH * HID * sizeof(float);               // 16 MB
  const size_t szP  = (size_t)BATCH * OUTP * sizeof(float);              // 8 MB
  __bf16* pOH = (__bf16*)(ws);
  __bf16* pHP = (__bf16*)(ws + szOH);
  __bf16* pPP = (__bf16*)(ws + szOH + szHP);
  float*  h0  = (float*)(ws + szOH + szHP + szPP);
  float*  h1  = (float*)((char*)h0 + szH);
  float*  p0  = (float*)((char*)h1 + szH);
  float*  p1  = (float*)((char*)p0 + szP);

  // Re-pack weights + zero initial states every call (deterministic, cheap).
  pack_oh_kernel<<<(KT_OH * NT_OH * 32 + 255) / 256, 256, 0, stream>>>(w_oh, pOH);
  pack_hp_kernel<<<(KT_HP * NT_HP * 32 + 255) / 256, 256, 0, stream>>>(w_hp, pHP);
  pack_pp_kernel<<<(KT_PP * NT_HP * 32 + 255) / 256, 256, 0, stream>>>(w_pp, pPP);
  zero_f32_kernel<<<2048, 256, 0, stream>>>(h0, BATCH * HID);
  zero_f32_kernel<<<1024, 256, 0, stream>>>(p0, BATCH * OUTP);

  // 8192 rows / 16 rows-per-wave / 4 waves-per-block = 128 blocks.
  rnn_fused_kernel<<<BATCH / 64, BLOCK_THREADS, 0, stream>>>(
      x, bias_h, bias_p, pOH, pHP, pPP, h0, h1, p0, p1, out);
}
